// GeoVOModel_87084756894236
// MI455X (gfx1250) — matrix operand order repaired and checked by reading, not compiled, not run on hardware
//
#include <hip/hip_runtime.h>
#include <hip/hip_bf16.h>
#include <math.h>

#define N_NODES 20000
#define N_EDGES 320000
#define N_TRI   100000
#define HEADS   4
#define HEAD_DIM 64
#define GRID_W 1216
#define IMG_W 1216.0f
#define IMG_H 352.0f

typedef __bf16 v16bf __attribute__((ext_vector_type(16)));
typedef __bf16 bf16x8 __attribute__((ext_vector_type(8)));
typedef float  v8f   __attribute__((ext_vector_type(8)));

__device__ __forceinline__ float gv_silu(float v) { return v / (1.0f + __expf(-v)); }

__device__ __forceinline__ unsigned gv_foi(float f) {
    unsigned u = __float_as_uint(f);
    return (u & 0x80000000u) ? ~u : (u | 0x80000000u);
}
__device__ __forceinline__ float gv_iof(unsigned u) {
    u = (u & 0x80000000u) ? (u & 0x7fffffffu) : ~u;
    return __uint_as_float(u);
}

// ---------------------------------------------------------------------------
// Wave-tile WMMA GEMM:  D[M,N] = act( A[M,K] * W[N,K]^T + bias )
// One wave computes a 32(M) x 64(N) tile: 2 A-fragments x 4 B-fragments,
// 8 x v_wmma_f32_16x16x32_bf16 per K-step. B fragments are contiguous-K rows
// of W, so each is a single 32B load; A fragments use the same contiguous-K
// order (lane half selects the K sub-chunk).
// GATHER: A row r, col k -> A[gidx[r*3 + k/256]*256 + k%256]  (f_tri fused)
// ---------------------------------------------------------------------------
template<bool GATHER, bool SILU, bool OUTF, bool OUTB>
__global__ void gv_gemm(const __bf16* __restrict__ A, const __bf16* __restrict__ W,
                        const float* __restrict__ bias,
                        float* __restrict__ Df, __bf16* __restrict__ Db,
                        const int* __restrict__ gidx, int M, int N, int K)
{
    int gtid = blockIdx.x * blockDim.x + threadIdx.x;
    int wave = gtid >> 5;
    int lane = threadIdx.x & 31;
    int NT = N >> 6;
    int MT = (M + 31) >> 5;
    if (wave >= MT * NT) return;
    int mt = wave / NT, nt = wave % NT;

    int m    = lane & 15;
    int ksel = lane >> 4;
    long row0 = (long)mt * 32 + m;
    long row1 = row0 + 16;
    int ncol0 = nt * 64;

    // Per-output-column B base pointers (constant across the K loop).
    const __bf16* wb[4];
    #pragma unroll
    for (int t = 0; t < 4; t++)
        wb[t] = W + (long)(ncol0 + t * 16 + m) * K + ksel * 16;

    v8f acc[2][4] = {};

    auto kstep = [&](const __bf16* a0, const __bf16* a1, int k) {
        union { v16bf v; bf16x8 h[2]; } af0, af1;
        af0.h[0] = *(const bf16x8*)(a0 + ksel * 16);
        af0.h[1] = *(const bf16x8*)(a0 + ksel * 16 + 8);
        af1.h[0] = *(const bf16x8*)(a1 + ksel * 16);
        af1.h[1] = *(const bf16x8*)(a1 + ksel * 16 + 8);
        __builtin_prefetch(a0 + 64, 0, 3);
        __builtin_prefetch(a1 + 64, 0, 3);
        #pragma unroll
        for (int t = 0; t < 4; t++) {
            v16bf bf = *(const v16bf*)(wb[t] + k);
            acc[0][t] = __builtin_amdgcn_wmma_f32_16x16x32_bf16(
                            false, af0.v, false, bf, (short)0, acc[0][t], false, false);
            acc[1][t] = __builtin_amdgcn_wmma_f32_16x16x32_bf16(
                            false, af1.v, false, bf, (short)0, acc[1][t], false, false);
        }
    };

    if (GATHER) {
        for (int kb = 0; kb < K; kb += 256) {
            const __bf16* a0 = A + (long)gidx[row0 * 3 + (kb >> 8)] * 256;
            const __bf16* a1 = A + (long)gidx[row1 * 3 + (kb >> 8)] * 256;
            #pragma unroll
            for (int kk = 0; kk < 256; kk += 32)
                kstep(a0 + kk, a1 + kk, kb + kk);
        }
    } else {
        const __bf16* a0 = A + row0 * K;
        const __bf16* a1 = A + row1 * K;
        for (int k = 0; k < K; k += 32)
            kstep(a0 + k, a1 + k, k);
    }

    int hsel = lane >> 4;
    #pragma unroll
    for (int t = 0; t < 4; t++) {
        int col = ncol0 + t * 16 + (lane & 15);
        float bv = bias ? bias[col] : 0.0f;
        #pragma unroll
        for (int half = 0; half < 2; half++) {
            #pragma unroll
            for (int r = 0; r < 8; r++) {
                long rw = (long)mt * 32 + half * 16 + r + hsel * 8;
                float v = acc[half][t][r] + bv;
                if (SILU) v = gv_silu(v);
                if (OUTF) Df[rw * N + col] = v;
                if (OUTB) Db[rw * N + col] = (__bf16)v;
            }
        }
    }
}

// ---------------------------------------------------------------------------
// f32 -> bf16 strided conversion (weight repack / activation concat)
// ---------------------------------------------------------------------------
__global__ void gv_cvt_bf16(__bf16* dst, int dstStride, const float* src, int srcStride,
                            int rows, int cols)
{
    long i = blockIdx.x * (long)blockDim.x + threadIdx.x;
    if (i >= (long)rows * cols) return;
    int r = (int)(i / cols), c = (int)(i % cols);
    dst[(long)r * dstStride + c] = (__bf16)src[(long)r * srcStride + c];
}

__global__ void gv_copy_f32(float* dst, const float* src, int n)
{
    int i = blockIdx.x * blockDim.x + threadIdx.x;
    if (i < n) dst[i] = src[i];
}

__global__ void gv_fill_u32(unsigned* p, unsigned v, int n)
{
    int i = blockIdx.x * blockDim.x + threadIdx.x;
    if (i < n) p[i] = v;
}

// ---------------------------------------------------------------------------
// Per-node positional MLP (3->32->64) + norm_uv; writes bf16 cols 256..319
// ---------------------------------------------------------------------------
__global__ void gv_node_pos(const float* __restrict__ kpts, const float* __restrict__ pt,
                            const float* __restrict__ w1, const float* __restrict__ b1,
                            const float* __restrict__ w2, const float* __restrict__ b2,
                            __bf16* __restrict__ nodex, float* __restrict__ normuv)
{
    int i = blockIdx.x * blockDim.x + threadIdx.x;
    if (i >= N_NODES) return;
    float u = kpts[2 * i] / IMG_W;
    float v = kpts[2 * i + 1] / IMG_H;
    normuv[2 * i] = u; normuv[2 * i + 1] = v;
    float pin[3] = {u, v, pt[3 * i + 2]};
    float h1[32];
    #pragma unroll
    for (int o = 0; o < 32; o++) {
        float s = b1[o];
        s += w1[o * 3 + 0] * pin[0] + w1[o * 3 + 1] * pin[1] + w1[o * 3 + 2] * pin[2];
        h1[o] = gv_silu(s);
    }
    for (int o = 0; o < 64; o++) {
        float s = b2[o];
        #pragma unroll
        for (int j = 0; j < 32; j++) s += w2[o * 32 + j] * h1[j];
        nodex[(long)i * 320 + 256 + o] = (__bf16)s;
    }
}

// ---------------------------------------------------------------------------
// Edge attention: wave per (edge, head) -> logit + ordered-uint atomic max
// ---------------------------------------------------------------------------
__global__ void gv_edge_logits(const int* __restrict__ ei, const float* __restrict__ nuv,
                               const float* __restrict__ big, const float* __restrict__ we,
                               const float* __restrict__ att,
                               float* __restrict__ logits, unsigned* __restrict__ mxenc)
{
    int gtid = blockIdx.x * blockDim.x + threadIdx.x;
    int wid = gtid >> 5, lane = gtid & 31;
    if (wid >= N_EDGES * HEADS) return;
    int e = wid >> 2, h = wid & 3;
    int s = ei[e], d = ei[N_EDGES + e];
    float rx = nuv[2 * d] - nuv[2 * s];
    float ry = nuv[2 * d + 1] - nuv[2 * s + 1];
    float dist = sqrtf(rx * rx + ry * ry);
    float sum = 0.0f;
    #pragma unroll
    for (int t = 0; t < 2; t++) {
        int c = h * 64 + lane + t * 32;
        float ee = rx * we[c * 3] + ry * we[c * 3 + 1] + dist * we[c * 3 + 2];
        float m = big[(long)s * 768 + 256 + c] + big[(long)d * 768 + 512 + c] + ee;
        m = m > 0.0f ? m : 0.2f * m;
        sum += m * att[c];
    }
    for (int o = 16; o; o >>= 1) sum += __shfl_down(sum, o);
    if (lane == 0) {
        logits[e * 4 + h] = sum;
        atomicMax(&mxenc[d * 4 + h], gv_foi(sum));
    }
}

__global__ void gv_edge_exp(const int* __restrict__ ei, const float* __restrict__ logits,
                            const unsigned* __restrict__ mxenc,
                            float* __restrict__ aexp, float* __restrict__ den)
{
    int i = blockIdx.x * blockDim.x + threadIdx.x;
    if (i >= N_EDGES * HEADS) return;
    int e = i >> 2, h = i & 3;
    int d = ei[N_EDGES + e];
    float a = __expf(logits[i] - gv_iof(mxenc[d * 4 + h]));
    aexp[i] = a;
    atomicAdd(&den[d * 4 + h], a);
}

__global__ void gv_edge_agg(const int* __restrict__ ei, const float* __restrict__ aexp,
                            const float* __restrict__ den, const float* __restrict__ big,
                            float* __restrict__ agg)
{
    long i = blockIdx.x * (long)blockDim.x + threadIdx.x;
    if (i >= (long)N_EDGES * 256) return;
    int e = (int)(i >> 8), cc = (int)(i & 255), h = cc >> 6;
    int s = ei[e], d = ei[N_EDGES + e];
    float alpha = aexp[e * 4 + h] / (den[d * 4 + h] + 1e-16f);
    atomicAdd(&agg[(long)d * 256 + cc], alpha * big[(long)s * 768 + 256 + cc]);
}

// ---------------------------------------------------------------------------
// LayerNorm + SiLU + residual ; one 256-thread block per node -> bf16
// ---------------------------------------------------------------------------
__global__ void gv_ln(const float* __restrict__ agg, const float* __restrict__ big,
                      const float* __restrict__ gbias, const float* __restrict__ g,
                      const float* __restrict__ b, __bf16* __restrict__ out)
{
    int i = blockIdx.x, c = threadIdx.x;
    float v = agg[(long)i * 256 + c] + gbias[c];
    float s = v, s2 = v * v;
    for (int o = 16; o; o >>= 1) { s += __shfl_down(s, o); s2 += __shfl_down(s2, o); }
    __shared__ float ps[8], ps2[8];
    int w = c >> 5, l = c & 31;
    if (l == 0) { ps[w] = s; ps2[w] = s2; }
    __syncthreads();
    if (c == 0) {
        float S = 0, S2 = 0;
        for (int k = 0; k < 8; k++) { S += ps[k]; S2 += ps2[k]; }
        float mu = S / 256.0f;
        ps[0] = mu; ps2[0] = S2 / 256.0f - mu * mu;
    }
    __syncthreads();
    float y = (v - ps[0]) * rsqrtf(ps2[0] + 1e-5f) * g[c] + b[c];
    out[(long)i * 256 + c] = (__bf16)(gv_silu(y) + big[(long)i * 768 + c]);
}

// ---------------------------------------------------------------------------
// Triangle heads: w_j = sigmoid(h.ww+b), n_j = normalize(tanh(h.nw+b))
// ---------------------------------------------------------------------------
__global__ void gv_heads(const __bf16* __restrict__ h, const float* __restrict__ ww,
                         const float* __restrict__ wb, const float* __restrict__ nw,
                         const float* __restrict__ nb,
                         float* __restrict__ nj, float* __restrict__ wj)
{
    int gtid = blockIdx.x * blockDim.x + threadIdx.x;
    int t = gtid >> 5, lane = gtid & 31;
    if (t >= N_TRI) return;
    int base = lane * 16;
    float sw = 0, s0 = 0, s1 = 0, s2 = 0;
    #pragma unroll
    for (int e = 0; e < 16; e++) {
        float hv = (float)h[(long)t * 512 + base + e];
        sw += hv * ww[base + e];
        s0 += hv * nw[base + e];
        s1 += hv * nw[512 + base + e];
        s2 += hv * nw[1024 + base + e];
    }
    for (int o = 16; o; o >>= 1) {
        sw += __shfl_down(sw, o); s0 += __shfl_down(s0, o);
        s1 += __shfl_down(s1, o); s2 += __shfl_down(s2, o);
    }
    if (lane == 0) {
        wj[t] = 1.0f / (1.0f + __expf(-(sw + wb[0])));
        float n0 = tanhf(s0 + nb[0]), n1 = tanhf(s1 + nb[1]), n2 = tanhf(s2 + nb[2]);
        float nrm = fmaxf(sqrtf(n0 * n0 + n1 * n1 + n2 * n2), 1e-12f);
        nj[t * 3 + 0] = n0 / nrm; nj[t * 3 + 1] = n1 / nrm; nj[t * 3 + 2] = n2 / nrm;
    }
}

// ---------------------------------------------------------------------------
// 3x3 SVD-based procrustes (Jacobi eigen of K^T K)
// ---------------------------------------------------------------------------
__device__ void gv_eig3(float A[3][3], float V[3][3])
{
    for (int i = 0; i < 3; i++) for (int j = 0; j < 3; j++) V[i][j] = (i == j) ? 1.0f : 0.0f;
    for (int sweep = 0; sweep < 12; sweep++) {
        for (int p = 0; p < 2; p++) for (int q = p + 1; q < 3; q++) {
            float apq = A[p][q];
            if (fabsf(apq) < 1e-24f) continue;
            float theta = (A[q][q] - A[p][p]) / (2.0f * apq);
            float tt = 1.0f / (fabsf(theta) + sqrtf(theta * theta + 1.0f));
            if (theta < 0.0f) tt = -tt;
            float c = rsqrtf(tt * tt + 1.0f), s = tt * c;
            for (int k = 0; k < 3; k++) {
                float akp = A[k][p], akq = A[k][q];
                A[k][p] = c * akp - s * akq; A[k][q] = s * akp + c * akq;
            }
            for (int k = 0; k < 3; k++) {
                float apk = A[p][k], aqk = A[q][k];
                A[p][k] = c * apk - s * aqk; A[q][k] = s * apk + c * aqk;
            }
            for (int k = 0; k < 3; k++) {
                float vkp = V[k][p], vkq = V[k][q];
                V[k][p] = c * vkp - s * vkq; V[k][q] = s * vkp + c * vkq;
            }
        }
    }
}

__device__ void gv_procrustes3(const float K[3][3], float R[3][3])
{
    float C[3][3], V[3][3], w[3];
    for (int i = 0; i < 3; i++) for (int j = 0; j < 3; j++)
        C[i][j] = K[0][i] * K[0][j] + K[1][i] * K[1][j] + K[2][i] * K[2][j];
    gv_eig3(C, V);
    w[0] = C[0][0]; w[1] = C[1][1]; w[2] = C[2][2];
    for (int a = 0; a < 2; a++) for (int b = a + 1; b < 3; b++)
        if (w[b] > w[a]) {
            float tw = w[a]; w[a] = w[b]; w[b] = tw;
            for (int k = 0; k < 3; k++) { float tv = V[k][a]; V[k][a] = V[k][b]; V[k][b] = tv; }
        }
    float U[3][3];
    for (int c = 0; c < 3; c++) {
        float inv = 1.0f / sqrtf(fmaxf(w[c], 1e-24f));
        for (int k = 0; k < 3; k++)
            U[k][c] = (K[k][0] * V[0][c] + K[k][1] * V[1][c] + K[k][2] * V[2][c]) * inv;
    }
    float det = K[0][0] * (K[1][1] * K[2][2] - K[1][2] * K[2][1])
              - K[0][1] * (K[1][0] * K[2][2] - K[1][2] * K[2][0])
              + K[0][2] * (K[1][0] * K[2][1] - K[1][1] * K[2][0]);
    float d3 = det >= 0.0f ? 1.0f : -1.0f;
    for (int i = 0; i < 3; i++) for (int j = 0; j < 3; j++)
        R[i][j] = V[i][0] * U[j][0] + V[i][1] * U[j][1] + d3 * V[i][2] * U[j][2];
}

__global__ void gv_tri_procrustes(const int* __restrict__ tri, const float* __restrict__ pt,
                                  const float* __restrict__ pm, const float* __restrict__ focal,
                                  const float* __restrict__ cx,
                                  float* __restrict__ Kj, float* __restrict__ xv)
{
    int t = blockIdx.x * blockDim.x + threadIdx.x;
    if (t >= N_TRI) return;
    float a[3][3], b[3][3];
    for (int i = 0; i < 3; i++) {
        int idx = tri[t * 3 + i];
        for (int j = 0; j < 3; j++) { a[i][j] = pt[idx * 3 + j]; b[i][j] = pm[idx * 3 + j]; }
    }
    for (int j = 0; j < 3; j++) {
        float ma = (a[0][j] + a[1][j] + a[2][j]) * (1.0f / 3.0f);
        float mb = (b[0][j] + b[1][j] + b[2][j]) * (1.0f / 3.0f);
        for (int i = 0; i < 3; i++) { a[i][j] -= ma; b[i][j] -= mb; }
    }
    float K[3][3];
    for (int j = 0; j < 3; j++) for (int k = 0; k < 3; k++) {
        K[j][k] = a[0][j] * b[0][k] + a[1][j] * b[1][k] + a[2][j] * b[2][k];
        Kj[(long)t * 9 + j * 3 + k] = K[j][k];
    }
    float R[3][3];
    gv_procrustes3(K, R);
    xv[t] = focal[0] * (R[0][2] / (R[2][2] + 1e-8f)) + cx[0];
}

// ---------------------------------------------------------------------------
// Voting (truncated Gaussian window; beyond +-40px contributions are < 1e-87)
// ---------------------------------------------------------------------------
__global__ void gv_vote(const float* __restrict__ xv, const float* __restrict__ wj,
                        float* __restrict__ voting)
{
    int t = blockIdx.x * blockDim.x + threadIdx.x;
    if (t >= N_TRI) return;
    float x = fminf(fmaxf(xv[t], -1e6f), 1e6f);
    float w = wj[t];
    int g0 = max(0, (int)floorf(x) - 40);
    int g1 = min(GRID_W - 1, (int)ceilf(x) + 40);
    for (int g = g0; g <= g1; g++) {
        float dd = x - (float)g;
        atomicAdd(&voting[g], w * __expf(-dd * dd * 0.125f));
    }
}

__global__ void gv_vote_softmax(const float* __restrict__ voting, float* __restrict__ xstar)
{
    __shared__ float sm[8], sm2[8];
    int c = threadIdx.x, w = c >> 5, l = c & 31;
    float mx = -3.4e38f;
    for (int g = c; g < GRID_W; g += 256) mx = fmaxf(mx, voting[g] * 10.0f);
    for (int o = 16; o; o >>= 1) mx = fmaxf(mx, __shfl_down(mx, o));
    if (l == 0) sm[w] = mx;
    __syncthreads();
    if (c == 0) { float M = sm[0]; for (int k = 1; k < 8; k++) M = fmaxf(M, sm[k]); sm[0] = M; }
    __syncthreads();
    float M = sm[0], se = 0, sg = 0;
    for (int g = c; g < GRID_W; g += 256) {
        float p = __expf(voting[g] * 10.0f - M);
        se += p; sg += p * (float)g;
    }
    for (int o = 16; o; o >>= 1) { se += __shfl_down(se, o); sg += __shfl_down(sg, o); }
    __syncthreads();
    if (l == 0) { sm[w] = se; sm2[w] = sg; }
    __syncthreads();
    if (c == 0) {
        float SE = 0, SG = 0;
        for (int k = 0; k < 8; k++) { SE += sm[k]; SG += sm2[k]; }
        xstar[0] = SG / SE;
    }
}

__global__ void gv_final_weights(const float* __restrict__ xv, const float* __restrict__ wj,
                                 const float* __restrict__ Kj, const float* __restrict__ xstar,
                                 float* __restrict__ fw, float* __restrict__ Ktot)
{
    int gtid = blockIdx.x * blockDim.x + threadIdx.x;
    int t = gtid, lane = threadIdx.x & 31;
    float f = 0.0f, loc[9];
    if (t < N_TRI) {
        float dx = xv[t] - xstar[0];
        f = wj[t] * __expf(-dx * dx * 0.125f);
        fw[t] = f;
        for (int q = 0; q < 9; q++) loc[q] = f * Kj[(long)t * 9 + q];
    } else {
        for (int q = 0; q < 9; q++) loc[q] = 0.0f;
    }
    for (int o = 16; o; o >>= 1)
        for (int q = 0; q < 9; q++) loc[q] += __shfl_down(loc[q], o);
    if (lane == 0)
        for (int q = 0; q < 9; q++) atomicAdd(&Ktot[q], loc[q]);
}

__global__ void gv_final_R(const float* __restrict__ Ktot, float* __restrict__ out)
{
    if (blockIdx.x | threadIdx.x) return;
    float K[3][3], R[3][3];
    for (int i = 0; i < 3; i++) for (int j = 0; j < 3; j++)
        K[i][j] = Ktot[i * 3 + j] + (i == j ? 1e-6f : 0.0f);
    gv_procrustes3(K, R);
    for (int i = 0; i < 3; i++) for (int j = 0; j < 3; j++) out[i * 3 + j] = R[i][j];
}

// ---------------------------------------------------------------------------
// Host launcher
// ---------------------------------------------------------------------------
static inline unsigned gv_blocks(long n, int bs) { return (unsigned)((n + bs - 1) / bs); }

extern "C" void kernel_launch(void* const* d_in, const int* in_sizes, int n_in,
                              void* d_out, int out_size, void* d_ws, size_t ws_size,
                              hipStream_t stream)
{
    const float* x      = (const float*)d_in[0];
    const float* kpts   = (const float*)d_in[1];
    const float* pt     = (const float*)d_in[2];
    const float* pm     = (const float*)d_in[3];
    const int*   ei     = (const int*)d_in[4];
    const int*   tri    = (const int*)d_in[5];
    const float* focal  = (const float*)d_in[6];
    const float* cx     = (const float*)d_in[7];
    const float* pos_w1 = (const float*)d_in[8];
    const float* pos_b1 = (const float*)d_in[9];
    const float* pos_w2 = (const float*)d_in[10];
    const float* pos_b2 = (const float*)d_in[11];
    const float* res_w  = (const float*)d_in[12];
    const float* res_b  = (const float*)d_in[13];
    const float* gat_wl = (const float*)d_in[14];
    const float* gat_bl = (const float*)d_in[15];
    const float* gat_wr = (const float*)d_in[16];
    const float* gat_br = (const float*)d_in[17];
    const float* gat_we = (const float*)d_in[18];
    const float* gat_att= (const float*)d_in[19];
    const float* gat_bias=(const float*)d_in[20];
    const float* ln_g   = (const float*)d_in[21];
    const float* ln_b   = (const float*)d_in[22];
    const float* proj_w = (const float*)d_in[23];
    const float* proj_b = (const float*)d_in[24];
    const float* tri_w1 = (const float*)d_in[25];
    const float* tri_b1 = (const float*)d_in[26];
    const float* tri_ww = (const float*)d_in[27];
    const float* tri_wb = (const float*)d_in[28];
    const float* tri_nw = (const float*)d_in[29];
    const float* tri_nb = (const float*)d_in[30];
    (void)in_sizes; (void)n_in; (void)out_size; (void)ws_size;

    char* wsp = (char*)d_ws;
    auto alloc = [&](size_t bytes) -> void* {
        void* p = (void*)wsp;
        wsp += (bytes + 255) & ~(size_t)255;
        return p;
    };
    __bf16* nodex_bf  = (__bf16*)alloc((size_t)N_NODES * 320 * 2);
    float*  normuv    = (float*) alloc((size_t)N_NODES * 2 * 4);
    __bf16* Wcat_bf   = (__bf16*)alloc((size_t)768 * 320 * 2);
    float*  biascat   = (float*) alloc(768 * 4);
    __bf16* projw_bf  = (__bf16*)alloc((size_t)256 * 256 * 2);
    __bf16* triw_bf   = (__bf16*)alloc((size_t)512 * 768 * 2);
    float*  big       = (float*) alloc((size_t)N_NODES * 768 * 4);
    float*  logits    = (float*) alloc((size_t)N_EDGES * 4 * 4);
    float*  aexp      = (float*) alloc((size_t)N_EDGES * 4 * 4);
    unsigned* mxenc   = (unsigned*)alloc((size_t)N_NODES * 4 * 4);
    float*  den       = (float*) alloc((size_t)N_NODES * 4 * 4);
    float*  agg       = (float*) alloc((size_t)N_NODES * 256 * 4);
    __bf16* act_bf    = (__bf16*)alloc((size_t)N_NODES * 256 * 2);
    __bf16* nodeout_bf= (__bf16*)alloc((size_t)N_NODES * 256 * 2);
    __bf16* h_bf      = (__bf16*)alloc((size_t)N_TRI * 512 * 2);
    float*  wj        = (float*) alloc((size_t)N_TRI * 4);
    float*  xv        = (float*) alloc((size_t)N_TRI * 4);
    float*  Kj        = (float*) alloc((size_t)N_TRI * 9 * 4);
    float*  voting    = (float*) alloc(GRID_W * 4);
    float*  xstar     = (float*) alloc(256);
    float*  Ktot      = (float*) alloc(9 * 4);

    float* out_R  = (float*)d_out;           // 9
    float* out_nj = out_R + 9;               // 300000
    float* out_fw = out_nj + (size_t)N_TRI*3;// 100000

    // --- zero init (graph-capture-safe) ---
    hipMemsetAsync(den,    0, (size_t)N_NODES * 4 * 4, stream);
    hipMemsetAsync(agg,    0, (size_t)N_NODES * 256 * 4, stream);
    hipMemsetAsync(voting, 0, GRID_W * 4, stream);
    hipMemsetAsync(Ktot,   0, 9 * 4, stream);
    gv_fill_u32<<<gv_blocks(N_NODES * 4, 256), 256, 0, stream>>>(mxenc, 0x007FFFFFu, N_NODES * 4); // foi(-inf)

    // --- precision repack: activations + weights to bf16 ---
    gv_cvt_bf16<<<gv_blocks((long)N_NODES * 256, 256), 256, 0, stream>>>(nodex_bf, 320, x, 256, N_NODES, 256);
    gv_node_pos<<<gv_blocks(N_NODES, 256), 256, 0, stream>>>(kpts, pt, pos_w1, pos_b1, pos_w2, pos_b2, nodex_bf, normuv);
    gv_cvt_bf16<<<gv_blocks(256 * 320, 256), 256, 0, stream>>>(Wcat_bf,             320, res_w,  320, 256, 320);
    gv_cvt_bf16<<<gv_blocks(256 * 320, 256), 256, 0, stream>>>(Wcat_bf + 256 * 320, 320, gat_wl, 320, 256, 320);
    gv_cvt_bf16<<<gv_blocks(256 * 320, 256), 256, 0, stream>>>(Wcat_bf + 512 * 320, 320, gat_wr, 320, 256, 320);
    gv_copy_f32<<<1, 256, 0, stream>>>(biascat,       res_b,  256);
    gv_copy_f32<<<1, 256, 0, stream>>>(biascat + 256, gat_bl, 256);
    gv_copy_f32<<<1, 256, 0, stream>>>(biascat + 512, gat_br, 256);
    gv_cvt_bf16<<<gv_blocks(256 * 256, 256), 256, 0, stream>>>(projw_bf, 256, proj_w, 256, 256, 256);
    gv_cvt_bf16<<<gv_blocks(512 * 768, 256), 256, 0, stream>>>(triw_bf,  768, tri_w1, 768, 512, 768);

    // --- GEMM 1: [identity | xl | xr] = node_x @ Wcat^T  (20000x320x768) ---
    {
        long waves = (long)(N_NODES / 32) * (768 / 64);
        gv_gemm<false,false,true,false><<<gv_blocks(waves * 32, 256), 256, 0, stream>>>(
            nodex_bf, Wcat_bf, biascat, big, nullptr, nullptr, N_NODES, 768, 320);
    }

    // --- edge attention (segment softmax via atomics) ---
    gv_edge_logits<<<gv_blocks((long)N_EDGES * HEADS * 32, 256), 256, 0, stream>>>(
        ei, normuv, big, gat_we, gat_att, logits, mxenc);
    gv_edge_exp<<<gv_blocks(N_EDGES * HEADS, 256), 256, 0, stream>>>(ei, logits, mxenc, aexp, den);
    gv_edge_agg<<<gv_blocks((long)N_EDGES * 256, 256), 256, 0, stream>>>(ei, aexp, den, big, agg);

    // --- LN + SiLU + residual ---
    gv_ln<<<N_NODES, 256, 0, stream>>>(agg, big, gat_bias, ln_g, ln_b, act_bf);

    // --- GEMM 2: node_out = act @ proj^T (20000x256x256) -> bf16 ---
    {
        long waves = (long)(N_NODES / 32) * (256 / 64);
        gv_gemm<false,false,false,true><<<gv_blocks(waves * 32, 256), 256, 0, stream>>>(
            act_bf, projw_bf, proj_b, nullptr, nodeout_bf, nullptr, N_NODES, 256, 256);
    }

    // --- GEMM 3 (dominant, fused gather + SiLU): h = silu(f_tri @ tri_w1^T) ---
    {
        long waves = (long)(N_TRI / 32) * (512 / 64);
        gv_gemm<true,true,false,true><<<gv_blocks(waves * 32, 256), 256, 0, stream>>>(
            nodeout_bf, triw_bf, tri_b1, nullptr, h_bf, tri, N_TRI, 512, 768);
    }

    // --- heads, procrustes, voting, consensus ---
    gv_heads<<<gv_blocks((long)N_TRI * 32, 256), 256, 0, stream>>>(h_bf, tri_ww, tri_wb, tri_nw, tri_nb, out_nj, wj);
    gv_tri_procrustes<<<gv_blocks(N_TRI, 256), 256, 0, stream>>>(tri, pt, pm, focal, cx, Kj, xv);
    gv_vote<<<gv_blocks(N_TRI, 256), 256, 0, stream>>>(xv, wj, voting);
    gv_vote_softmax<<<1, 256, 0, stream>>>(voting, xstar);
    gv_final_weights<<<gv_blocks(N_TRI, 256), 256, 0, stream>>>(xv, wj, Kj, xstar, out_fw, Ktot);
    gv_final_R<<<1, 32, 0, stream>>>(Ktot, out_R);
}
